// CausalMultiHeadSelfAttention_16303695855889
// MI455X (gfx1250) — matrix operand order repaired and checked by reading, not compiled, not run on hardware
//
#include <hip/hip_runtime.h>

typedef _Float16 v16h __attribute__((ext_vector_type(16)));
typedef _Float16 v8h  __attribute__((ext_vector_type(8)));
typedef float    v8f  __attribute__((ext_vector_type(8)));

constexpr int Bsz  = 4;
constexpr int Tlen = 2048;
constexpr int E    = 1024;
constexpr int NH   = 16;
constexpr int HD   = 64;
constexpr int Mrows = Bsz * Tlen; // 8192

// softmax(s / 8) computed in base-2: exp2(s * 0.125 * log2(e))
#define SM_SCALE2 0.1803368801111904f

union FragH { v16h v; v8h h[2]; };

__device__ __forceinline__ v16h load_frag(const _Float16* p0, const _Float16* p1) {
  FragH f;
  f.h[0] = *(const v8h*)p0;
  f.h[1] = *(const v8h*)p1;
  return f.v;
}

__device__ __forceinline__ v8f wmma_f16(v16h a, v16h b, v8f c) {
  return __builtin_amdgcn_wmma_f32_16x16x32_f16(false, a, false, b, (short)0, c,
                                                false, false);
}

// ---------------------------------------------------------------------------
// Kernel 1: RoPE rotation + f32 -> f16 conversion of activations.
// ---------------------------------------------------------------------------
__global__ __launch_bounds__(256) void rope_prep(const float* __restrict__ x,
                                                 _Float16* __restrict__ rx_h,
                                                 _Float16* __restrict__ x_h) {
  int p = blockIdx.x * blockDim.x + threadIdx.x; // pair index
  if (p >= Mrows * (E / 2)) return;
  int m = p / (E / 2);
  int i = p - m * (E / 2);
  int t = m & (Tlen - 1);
  float theta = exp2f(-(float)(2 * i) * (13.287712379549449f / (float)E));
  float ang = (float)t * theta;
  float s = sinf(ang), c = cosf(ang);
  float x0 = x[(size_t)m * E + 2 * i];
  float x1 = x[(size_t)m * E + 2 * i + 1];
  x_h[(size_t)m * E + 2 * i]      = (_Float16)x0;
  x_h[(size_t)m * E + 2 * i + 1]  = (_Float16)x1;
  rx_h[(size_t)m * E + 2 * i]     = (_Float16)(x0 * c - x1 * s);
  rx_h[(size_t)m * E + 2 * i + 1] = (_Float16)(x1 * c + x0 * s);
}

// ---------------------------------------------------------------------------
// Kernel 2: weight f32 -> f16 conversion.
// ---------------------------------------------------------------------------
__global__ __launch_bounds__(256) void cvt_f32_f16(const float* __restrict__ src,
                                                   _Float16* __restrict__ dst, int n) {
  int i = blockIdx.x * blockDim.x + threadIdx.x;
  if (i < n) dst[i] = (_Float16)src[i];
}

// ---------------------------------------------------------------------------
// Kernel 3: QKV projections (grid.z: 0=Q, 1=K, 2=V-transposed).
// Wave computes a 32x64 tile (2 A-frags x 4 B-frags -> 8 WMMAs / k-step),
// reusing every B fragment twice. Q,K stored [b,h,t,d]; V stored [b,h,d,t].
// ---------------------------------------------------------------------------
__global__ __launch_bounds__(256) void qkv_gemm(
    const _Float16* __restrict__ rx_h, const _Float16* __restrict__ x_h,
    const _Float16* __restrict__ Wq_h, const _Float16* __restrict__ Wk_h,
    const _Float16* __restrict__ Wv_h,
    _Float16* __restrict__ Qd, _Float16* __restrict__ Kd, _Float16* __restrict__ Vt) {
  const int w    = threadIdx.x >> 5;
  const int lane = threadIdx.x & 31;
  const int lo   = lane & 15;
  const int hi8  = (lane >> 4) << 3;   // A-fragment K sub-offset (0 or 8)
  const int klo  = (lane >> 4) << 4;   // unused (0 or 16); B fragment uses hi8 like A
  const int m0   = blockIdx.x * 256 + w * 32;
  const int n0   = blockIdx.y * 64;
  const int mode = blockIdx.z;

  const _Float16* A = (mode == 2) ? x_h : rx_h;
  const _Float16* W = (mode == 0) ? Wq_h : (mode == 1) ? Wk_h : Wv_h;

  const size_t ar0 = (size_t)(m0 + lo) * E;
  const size_t ar1 = (size_t)(m0 + 16 + lo) * E;
  v8f acc[2][4] = {};

  for (int k = 0; k < E; k += 32) {
    v16h a0 = load_frag(A + ar0 + k + hi8, A + ar0 + k + 16 + hi8);
    v16h a1 = load_frag(A + ar1 + k + hi8, A + ar1 + k + 16 + hi8);
#pragma unroll
    for (int tl = 0; tl < 4; ++tl) {
      const _Float16* wp = W + (size_t)(n0 + tl * 16 + lo) * E + k + hi8;
      v16h b = load_frag(wp, wp + 16);
      acc[0][tl] = wmma_f16(a0, b, acc[0][tl]);
      acc[1][tl] = wmma_f16(a1, b, acc[1][tl]);
    }
  }

#pragma unroll
  for (int mi = 0; mi < 2; ++mi) {
#pragma unroll
    for (int tl = 0; tl < 4; ++tl) {
      int n = n0 + tl * 16 + lo;
      int h = n >> 6, d = n & 63;
#pragma unroll
      for (int v = 0; v < 8; ++v) {
        int m = m0 + mi * 16 + v + hi8;  // C-layout: row = v + 8*(lane>=16)
        int b = m >> 11, tt = m & (Tlen - 1);
        _Float16 val = (_Float16)acc[mi][tl][v];
        if (mode == 2)
          Vt[((size_t)(b * NH + h) * HD + d) * Tlen + tt] = val;
        else if (mode == 1)
          Kd[((size_t)(b * NH + h) * Tlen + tt) * HD + d] = val;
        else
          Qd[((size_t)(b * NH + h) * Tlen + tt) * HD + d] = val;
      }
    }
  }
}

// ---------------------------------------------------------------------------
// Kernel 4: causal flash attention. One wave owns a 16-row Q tile; key tiles
// of 64 (P tile = two 16x32 A-fragments -> 16 WMMAs per softmax pass).
// Online softmax in base-2 with 16-lane shuffle reductions; mask compares are
// skipped on fully-unmasked tiles (wave-uniform branch).
// ---------------------------------------------------------------------------
__global__ __launch_bounds__(128) void flash_attn(
    const _Float16* __restrict__ Qd, const _Float16* __restrict__ Kd,
    const _Float16* __restrict__ Vt, _Float16* __restrict__ O) {
  __shared__ __align__(16) _Float16 pt[4][16][64];

  const int w    = threadIdx.x >> 5;
  const int lane = threadIdx.x & 31;
  const int lo   = lane & 15;
  const int hi8  = (lane >> 4) << 3;
  const int klo  = (lane >> 4) << 4;
  const int h    = blockIdx.y, b = blockIdx.z;
  const int q0   = blockIdx.x * 64 + w * 16;

  const _Float16* Qb = Qd + (size_t)(b * NH + h) * Tlen * HD;
  const _Float16* Kb = Kd + (size_t)(b * NH + h) * Tlen * HD;
  const _Float16* Vb = Vt + (size_t)(b * NH + h) * HD * Tlen;

  const size_t qoff = (size_t)(q0 + lo) * HD;
  v16h aq0 = load_frag(Qb + qoff + hi8,      Qb + qoff + 16 + hi8); // d 0..31
  v16h aq1 = load_frag(Qb + qoff + 32 + hi8, Qb + qoff + 48 + hi8); // d 32..63

  float mrow[8], lrow[8];
  v8f acc[4] = {};
#pragma unroll
  for (int v = 0; v < 8; ++v) { mrow[v] = -1e30f; lrow[v] = 0.f; }

  const int qLast = q0 + 15;
  for (int j0 = 0; j0 <= qLast; j0 += 64) {
    // speculative prefetch of the next key tile (non-faulting, RT hint)
    __builtin_prefetch((const void*)(Kb + (size_t)(j0 + 64 + lane) * HD), 0, 1);
    __builtin_prefetch((const void*)(Kb + (size_t)(j0 + 96 + lane) * HD), 0, 1);
    __builtin_prefetch((const void*)(Vb + (size_t)lane * Tlen + j0 + 64), 0, 1);
    __builtin_prefetch((const void*)(Vb + (size_t)(32 + lane) * Tlen + j0 + 64), 0, 1);

    // --- scores: Q(16x64) x K^T(64x64) as four 16-key subtiles -------------
    v8f s[4];
#pragma unroll
    for (int jj = 0; jj < 4; ++jj) {
      const _Float16* kp = Kb + (size_t)(j0 + jj * 16 + lo) * HD; // B col = key
      v16h bka = load_frag(kp + hi8,      kp + 16 + hi8);         // d lo-half
      v16h bkb = load_frag(kp + 32 + hi8, kp + 48 + hi8);         // d hi-half
      v8f z = {};
      z = wmma_f16(aq0, bka, z);
      s[jj] = wmma_f16(aq1, bkb, z);
    }

    // --- scale, causal mask, online softmax (base-2) ----------------------
    const bool full = (j0 + 63 <= q0); // whole tile unmasked for all rows
#pragma unroll
    for (int v = 0; v < 8; ++v) {
      const int row = q0 + v + hi8;
      float t[4];
#pragma unroll
      for (int jj = 0; jj < 4; ++jj) t[jj] = s[jj][v] * SM_SCALE2;
      if (!full) {
#pragma unroll
        for (int jj = 0; jj < 4; ++jj)
          if (j0 + jj * 16 + lo > row) t[jj] = -1e30f;
      }
      float rm = fmaxf(fmaxf(t[0], t[1]), fmaxf(t[2], t[3]));
      rm = fmaxf(rm, __shfl_xor(rm, 1, 32));
      rm = fmaxf(rm, __shfl_xor(rm, 2, 32));
      rm = fmaxf(rm, __shfl_xor(rm, 4, 32));
      rm = fmaxf(rm, __shfl_xor(rm, 8, 32));
      float mnew  = fmaxf(mrow[v], rm);
      float alpha = __builtin_amdgcn_exp2f(mrow[v] - mnew);
      mrow[v] = mnew;
      float e[4];
#pragma unroll
      for (int jj = 0; jj < 4; ++jj) e[jj] = __builtin_amdgcn_exp2f(t[jj] - mnew);
      float rs = (e[0] + e[1]) + (e[2] + e[3]);
      rs += __shfl_xor(rs, 1, 32);
      rs += __shfl_xor(rs, 2, 32);
      rs += __shfl_xor(rs, 4, 32);
      rs += __shfl_xor(rs, 8, 32);
      lrow[v] = lrow[v] * alpha + rs;
#pragma unroll
      for (int tl = 0; tl < 4; ++tl) acc[tl][v] *= alpha;
#pragma unroll
      for (int jj = 0; jj < 4; ++jj)
        pt[w][v + hi8][jj * 16 + lo] = (_Float16)e[jj];
    }
    asm volatile("s_wait_dscnt 0" ::: "memory"); // wave-private tile; DS in-order

    // P (16x64 f16) as two A-fragments from LDS
    v16h ap0 = load_frag(&pt[w][lo][hi8],      &pt[w][lo][16 + hi8]);
    v16h ap1 = load_frag(&pt[w][lo][32 + hi8], &pt[w][lo][48 + hi8]);

    // --- acc += P(16x64) x V(64x64); Vt rows contiguous in j --------------
#pragma unroll
    for (int tl = 0; tl < 4; ++tl) {
      const _Float16* vp = Vb + (size_t)(tl * 16 + lo) * Tlen + j0;
      v16h bv0 = load_frag(vp + hi8,      vp + 16 + hi8);      // j0..j0+31
      v16h bv1 = load_frag(vp + 32 + hi8, vp + 48 + hi8);      // j0+32..j0+63
      acc[tl] = wmma_f16(ap0, bv0, acc[tl]);
      acc[tl] = wmma_f16(ap1, bv1, acc[tl]);
    }
  }

  // --- normalize, write O[B*T, E] f16 -------------------------------------
#pragma unroll
  for (int tl = 0; tl < 4; ++tl) {
    int col = h * HD + tl * 16 + lo;
#pragma unroll
    for (int v = 0; v < 8; ++v) {
      int row = q0 + v + hi8;
      O[(size_t)(b * Tlen + row) * E + col] = (_Float16)(acc[tl][v] / lrow[v]);
    }
  }
}

// ---------------------------------------------------------------------------
// Kernel 5: final projection Y = O @ Wo.T + bo (f32 output to d_out).
// Same 32x64-per-wave tiling as qkv_gemm.
// ---------------------------------------------------------------------------
__global__ __launch_bounds__(256) void out_gemm(
    const _Float16* __restrict__ O, const _Float16* __restrict__ Wo_h,
    const float* __restrict__ bo, float* __restrict__ Y) {
  const int w    = threadIdx.x >> 5;
  const int lane = threadIdx.x & 31;
  const int lo   = lane & 15;
  const int hi8  = (lane >> 4) << 3;
  const int klo  = (lane >> 4) << 4;
  const int m0   = blockIdx.x * 256 + w * 32;
  const int n0   = blockIdx.y * 64;

  const size_t ar0 = (size_t)(m0 + lo) * E;
  const size_t ar1 = (size_t)(m0 + 16 + lo) * E;
  v8f acc[2][4] = {};
  for (int k = 0; k < E; k += 32) {
    v16h a0 = load_frag(O + ar0 + k + hi8, O + ar0 + k + 16 + hi8);
    v16h a1 = load_frag(O + ar1 + k + hi8, O + ar1 + k + 16 + hi8);
#pragma unroll
    for (int tl = 0; tl < 4; ++tl) {
      const _Float16* wp = Wo_h + (size_t)(n0 + tl * 16 + lo) * E + k + hi8;
      v16h b = load_frag(wp, wp + 16);
      acc[0][tl] = wmma_f16(a0, b, acc[0][tl]);
      acc[1][tl] = wmma_f16(a1, b, acc[1][tl]);
    }
  }
#pragma unroll
  for (int mi = 0; mi < 2; ++mi) {
#pragma unroll
    for (int tl = 0; tl < 4; ++tl) {
      int n = n0 + tl * 16 + lo;
      float bias = bo[n];
#pragma unroll
      for (int v = 0; v < 8; ++v) {
        Y[(size_t)(m0 + mi * 16 + v + hi8) * E + n] = acc[mi][tl][v] + bias;
      }
    }
  }
}

// ---------------------------------------------------------------------------
extern "C" void kernel_launch(void* const* d_in, const int* in_sizes, int n_in,
                              void* d_out, int out_size, void* d_ws, size_t ws_size,
                              hipStream_t stream) {
  const float* x  = (const float*)d_in[0];
  const float* Wq = (const float*)d_in[1];
  const float* Wk = (const float*)d_in[2];
  const float* Wv = (const float*)d_in[3];
  const float* Wo = (const float*)d_in[4];
  const float* bo = (const float*)d_in[5];
  float* Y = (float*)d_out;

  char* ws = (char*)d_ws;
  size_t off = 0;
  auto take = [&](size_t bytes) -> void* {
    void* p = ws + off;
    off += (bytes + 255) & ~(size_t)255;
    return p;
  };
  _Float16* rx_h = (_Float16*)take((size_t)Mrows * E * 2);
  _Float16* x_h  = (_Float16*)take((size_t)Mrows * E * 2);
  _Float16* Wq_h = (_Float16*)take((size_t)E * E * 2);
  _Float16* Wk_h = (_Float16*)take((size_t)E * E * 2);
  _Float16* Wv_h = (_Float16*)take((size_t)E * E * 2);
  _Float16* Wo_h = (_Float16*)take((size_t)E * E * 2);
  _Float16* Qd   = (_Float16*)take((size_t)Mrows * E * 2);
  _Float16* Kd   = (_Float16*)take((size_t)Mrows * E * 2);
  _Float16* Vt   = (_Float16*)take((size_t)Mrows * E * 2);
  _Float16* O    = (_Float16*)take((size_t)Mrows * E * 2);

  // 1) RoPE + activation conversion
  {
    int n = Mrows * (E / 2);
    rope_prep<<<(n + 255) / 256, 256, 0, stream>>>(x, rx_h, x_h);
  }
  // 2) Weight conversions
  {
    int n = E * E;
    int g = (n + 255) / 256;
    cvt_f32_f16<<<g, 256, 0, stream>>>(Wq, Wq_h, n);
    cvt_f32_f16<<<g, 256, 0, stream>>>(Wk, Wk_h, n);
    cvt_f32_f16<<<g, 256, 0, stream>>>(Wv, Wv_h, n);
    cvt_f32_f16<<<g, 256, 0, stream>>>(Wo, Wo_h, n);
  }
  // 3) Q/K/V projections (z = 0/1/2)
  qkv_gemm<<<dim3(Mrows / 256, E / 64, 3), 256, 0, stream>>>(
      rx_h, x_h, Wq_h, Wk_h, Wv_h, Qd, Kd, Vt);
  // 4) Causal flash attention
  flash_attn<<<dim3(Tlen / 64, NH, Bsz), 128, 0, stream>>>(Qd, Kd, Vt, O);
  // 5) Output projection + bias
  out_gemm<<<dim3(Mrows / 256, E / 64), 256, 0, stream>>>(O, Wo_h, bo, Y);
}
